// GraphAttentionPooling_20074677142125
// MI455X (gfx1250) — matrix operand order repaired and checked by reading, not compiled, not run on hardware
//
#include <hip/hip_runtime.h>

// ---------------------------------------------------------------------------
// GraphAttentionPooling for MI455X (gfx1250, wave32, WMMA)
//
//   Q = X Wq, K = X Wk, V = X Wv     (one fused 12288x512x256 f16 WMMA GEMM)
//   S = Q K^T / 16 block-diagonal (batch sorted -> contiguous graph segments)
//   out[g] = c^T V_g,  c_j = sum_i exp(S_ij - m_i)/Z_i   (flash-style stats)
// ---------------------------------------------------------------------------

typedef __attribute__((ext_vector_type(16))) _Float16 v16h;
typedef __attribute__((ext_vector_type(8)))  _Float16 v8h;
typedef __attribute__((ext_vector_type(8)))  float    v8f;

#define N_NODES   12288
#define N_FEAT    256
#define KEY_DIM   128
#define N_GRAPHS  192
#define NCOLS     512          // [Wq | Wk | Wv] concatenated columns
#define MAXN      256          // max nodes per graph handled (mean 64, +24 sigma safe)
#define AS_STRIDE 40           // LDS row stride (halves) for GEMM tiles (bank spread)

// ---------------------------------------------------------------------------
// Kernel 0: pack weights -> f16, transposed to n-major: WhT[n][k] = W[k][n]
// ---------------------------------------------------------------------------
__global__ __launch_bounds__(256)
void k_build_wt(const float* __restrict__ Wq, const float* __restrict__ Wk,
                const float* __restrict__ Wv, _Float16* __restrict__ WhT) {
  const int n = blockIdx.x;      // 0..511  (scalar branch)
  const int k = threadIdx.x;     // 0..255
  float v;
  if (n < KEY_DIM)             v = Wq[k * KEY_DIM + n];
  else if (n < 2 * KEY_DIM)    v = Wk[k * KEY_DIM + (n - KEY_DIM)];
  else                         v = Wv[k * N_FEAT  + (n - 2 * KEY_DIM)];
  WhT[n * N_FEAT + k] = (_Float16)v;
}

// ---------------------------------------------------------------------------
// Kernel 1: fused projection GEMM  C[12288,512] = X[12288,256] @ W[256,512]
// 128x128 tile per block, 8 waves in 2x4 grid, each wave 4x2 WMMA tiles.
// blockIdx.y == 0 -> Qh (f16), == 1 -> Kh (f16), >= 2 -> Vf (f32)
// ---------------------------------------------------------------------------
__global__ __launch_bounds__(256)
void k_proj_gemm(const float* __restrict__ X, const _Float16* __restrict__ WhT,
                 _Float16* __restrict__ Qh, _Float16* __restrict__ Kh,
                 float* __restrict__ Vf) {
  __shared__ _Float16 As[128 * AS_STRIDE];   // A tile 128x32 (f16, converted)
  __shared__ _Float16 Bs[128 * AS_STRIDE];   // B tile 128(n) x 32(k), n-major

  const int tid   = threadIdx.x;
  const int lane  = tid & 31;
  const int wid   = tid >> 5;
  const int hi    = lane >> 4;       // half-wave select
  const int l15   = lane & 15;
  const int waveM = wid >> 2;        // 0..1  (64 rows each)
  const int waveN = wid & 3;         // 0..3  (32 cols each)
  const int mBase = blockIdx.x * 128;
  const int nBase = blockIdx.y * 128;

  const int ldRow = tid >> 1;        // 0..127
  const int ldCol = (tid & 1) * 16;  // 0 or 16 (halves)

  v8f acc[4][2];
#pragma unroll
  for (int tm = 0; tm < 4; ++tm)
#pragma unroll
    for (int tn = 0; tn < 2; ++tn)
      acc[tm][tn] = (v8f){0.f, 0.f, 0.f, 0.f, 0.f, 0.f, 0.f, 0.f};

  for (int k0 = 0; k0 < N_FEAT; k0 += 32) {
    // ---- global loads (A: fp32 -> f16 convert; B: already f16, n-major) ----
    const float* xp = X + (size_t)(mBase + ldRow) * N_FEAT + k0 + ldCol;
    float4 f0 = *(const float4*)(xp + 0);
    float4 f1 = *(const float4*)(xp + 4);
    float4 f2 = *(const float4*)(xp + 8);
    float4 f3 = *(const float4*)(xp + 12);
    v8h a0, a1;
    a0[0]=(_Float16)f0.x; a0[1]=(_Float16)f0.y; a0[2]=(_Float16)f0.z; a0[3]=(_Float16)f0.w;
    a0[4]=(_Float16)f1.x; a0[5]=(_Float16)f1.y; a0[6]=(_Float16)f1.z; a0[7]=(_Float16)f1.w;
    a1[0]=(_Float16)f2.x; a1[1]=(_Float16)f2.y; a1[2]=(_Float16)f2.z; a1[3]=(_Float16)f2.w;
    a1[4]=(_Float16)f3.x; a1[5]=(_Float16)f3.y; a1[6]=(_Float16)f3.z; a1[7]=(_Float16)f3.w;
    const _Float16* wp = WhT + (size_t)(nBase + ldRow) * N_FEAT + k0 + ldCol;
    v8h b0 = *(const v8h*)(wp + 0);
    v8h b1 = *(const v8h*)(wp + 8);

    __syncthreads();   // previous iteration's fragment reads done
    *(v8h*)(As + ldRow * AS_STRIDE + ldCol + 0) = a0;
    *(v8h*)(As + ldRow * AS_STRIDE + ldCol + 8) = a1;
    *(v8h*)(Bs + ldRow * AS_STRIDE + ldCol + 0) = b0;
    *(v8h*)(Bs + ldRow * AS_STRIDE + ldCol + 8) = b1;
    __syncthreads();

    // ---- B fragments: lane = col, 16 contiguous halves (K = hi*16 + e) ----
    v16h bfrag[2];
#pragma unroll
    for (int tn = 0; tn < 2; ++tn) {
      const _Float16* p = Bs + (waveN * 32 + tn * 16 + l15) * AS_STRIDE + hi * 16;
      v8h x0 = *(const v8h*)(p + 0);
      v8h x1 = *(const v8h*)(p + 8);
#pragma unroll
      for (int i = 0; i < 8; ++i) { bfrag[tn][i] = x0[i]; bfrag[tn][i + 8] = x1[i]; }
    }
    // ---- A fragments: lane = row, 16 contiguous halves (K = hi*16 + e) ----
#pragma unroll
    for (int tm = 0; tm < 4; ++tm) {
      const _Float16* p = As + (waveM * 64 + tm * 16 + l15) * AS_STRIDE;
      v8h x0 = *(const v8h*)(p + hi * 16);
      v8h x1 = *(const v8h*)(p + hi * 16 + 8);
      v16h afrag;
#pragma unroll
      for (int i = 0; i < 8; ++i) { afrag[i] = x0[i]; afrag[i + 8] = x1[i]; }
#pragma unroll
      for (int tn = 0; tn < 2; ++tn)
        acc[tm][tn] = __builtin_amdgcn_wmma_f32_16x16x32_f16(
            false, afrag, false, bfrag[tn], (short)0, acc[tm][tn], false, false);
    }
  }

  // ---- epilogue: destination decided entirely by blockIdx.y (scalar branch,
  //      no per-lane exec-mask routing) ----
  const int rowB = mBase + waveM * 64 + hi * 8;       // + tm*16 + r
  const int colW = waveN * 32 + l15;                  // + tn*16, local to 128-tile

  if (nBase == 0) {                                   // columns 0..127 -> Qh
#pragma unroll
    for (int tm = 0; tm < 4; ++tm)
#pragma unroll
      for (int tn = 0; tn < 2; ++tn)
#pragma unroll
        for (int r = 0; r < 8; ++r)
          Qh[(size_t)(rowB + tm * 16 + r) * KEY_DIM + colW + tn * 16] =
              (_Float16)acc[tm][tn][r];
  } else if (nBase == 128) {                          // columns 128..255 -> Kh
#pragma unroll
    for (int tm = 0; tm < 4; ++tm)
#pragma unroll
      for (int tn = 0; tn < 2; ++tn)
#pragma unroll
        for (int r = 0; r < 8; ++r)
          Kh[(size_t)(rowB + tm * 16 + r) * KEY_DIM + colW + tn * 16] =
              (_Float16)acc[tm][tn][r];
  } else {                                            // columns 256..511 -> Vf
    const int vcol = (nBase - 256) + colW;
#pragma unroll
    for (int tm = 0; tm < 4; ++tm)
#pragma unroll
      for (int tn = 0; tn < 2; ++tn)
#pragma unroll
        for (int r = 0; r < 8; ++r)
          Vf[(size_t)(rowB + tm * 16 + r) * N_FEAT + vcol + tn * 16] = acc[tm][tn][r];
  }
}

// ---------------------------------------------------------------------------
// Kernel 2: per-graph block attention + pooling
// ---------------------------------------------------------------------------
__device__ __forceinline__ int lower_bound_i(const int* __restrict__ a, int n, int v) {
  int lo = 0, hiB = n;
  while (lo < hiB) {
    int mid = (lo + hiB) >> 1;
    if (a[mid] < v) lo = mid + 1; else hiB = mid;
  }
  return lo;
}

// 16x16 score tile (ti,tj) of S_g = Q_g K_g^T ; fragments straight from L2.
__device__ __forceinline__ v8f score_tile(const _Float16* __restrict__ Qh,
                                          const _Float16* __restrict__ Kh,
                                          int s, int ti, int tj, int l15, int hi) {
  const _Float16* qb = Qh + (size_t)(s + ti * 16 + l15) * KEY_DIM;
  const _Float16* kb = Kh + (size_t)(s + tj * 16 + l15) * KEY_DIM;
  v8f acc = (v8f){0.f, 0.f, 0.f, 0.f, 0.f, 0.f, 0.f, 0.f};
#pragma unroll
  for (int kk = 0; kk < KEY_DIM; kk += 32) {
    v8h a0 = *(const v8h*)(qb + kk + hi * 16);
    v8h a1 = *(const v8h*)(qb + kk + hi * 16 + 8);
    v8h b0 = *(const v8h*)(kb + kk + hi * 16);
    v8h b1 = *(const v8h*)(kb + kk + hi * 16 + 8);
    v16h a, b;
#pragma unroll
    for (int i = 0; i < 8; ++i) { a[i] = a0[i]; a[i + 8] = a1[i]; b[i] = b0[i]; b[i + 8] = b1[i]; }
    acc = __builtin_amdgcn_wmma_f32_16x16x32_f16(false, a, false, b, (short)0, acc,
                                                 false, false);
  }
  return acc;
}

__global__ __launch_bounds__(256)
void k_attn_pool(const _Float16* __restrict__ Qh, const _Float16* __restrict__ Kh,
                 const float* __restrict__ Vf, const int* __restrict__ batch,
                 float* __restrict__ out) {
  __shared__ float mArr[MAXN];   // row max
  __shared__ float zArr[MAXN];   // 1/Z after pass A
  __shared__ float cArr[MAXN];   // column weights

  const int g   = blockIdx.x;
  const int tid = threadIdx.x;

  const int s = lower_bound_i(batch, N_NODES, g);
  const int e = lower_bound_i(batch, N_NODES, g + 1);
  int n = e - s;
  if (n > MAXN) n = MAXN;
  if (n <= 0) { out[(size_t)g * N_FEAT + tid] = 0.f; return; }

  const int nt   = (n + 15) >> 4;
  const int lane = tid & 31;
  const int wid  = tid >> 5;
  const int hi   = lane >> 4;
  const int l15  = lane & 15;
  const float scale = 0.0625f;           // 1/sqrt(256)

  // ---- Pass A: per-row online (max, sumexp); row tiles round-robin over waves
  for (int ti = wid; ti < nt; ti += 8) {
    float mrun[8], lrun[8];
#pragma unroll
    for (int r = 0; r < 8; ++r) { mrun[r] = -1e30f; lrun[r] = 0.f; }
    for (int tj = 0; tj < nt; ++tj) {
      v8f acc = score_tile(Qh, Kh, s, ti, tj, l15, hi);
      const bool cv = (tj * 16 + l15) < n;   // column valid
#pragma unroll
      for (int r = 0; r < 8; ++r) {
        const float sv = acc[r] * scale;
        float sm = cv ? sv : -1e30f;
        sm = fmaxf(sm, __shfl_xor(sm, 1));
        sm = fmaxf(sm, __shfl_xor(sm, 2));
        sm = fmaxf(sm, __shfl_xor(sm, 4));
        sm = fmaxf(sm, __shfl_xor(sm, 8));
        const float nm = fmaxf(mrun[r], sm);
        float p = cv ? __expf(sv - nm) : 0.f;
        p += __shfl_xor(p, 1);
        p += __shfl_xor(p, 2);
        p += __shfl_xor(p, 4);
        p += __shfl_xor(p, 8);
        lrun[r] = lrun[r] * __expf(mrun[r] - nm) + p;
        mrun[r] = nm;
      }
    }
    if (l15 == 0) {
#pragma unroll
      for (int r = 0; r < 8; ++r) {
        const int row = ti * 16 + hi * 8 + r;
        if (row < n) { mArr[row] = mrun[r]; zArr[row] = lrun[r]; }
      }
    }
  }
  __syncthreads();
  for (int i = tid; i < n; i += 256) zArr[i] = 1.f / zArr[i];
  __syncthreads();

  // ---- Pass B: column weights c_j = sum_i exp(s_ij - m_i)/Z_i
  for (int tj = wid; tj < nt; tj += 8) {
    float csum = 0.f;
    for (int ti = 0; ti < nt; ++ti) {
      v8f acc = score_tile(Qh, Kh, s, ti, tj, l15, hi);
#pragma unroll
      for (int r = 0; r < 8; ++r) {
        const int row = ti * 16 + hi * 8 + r;
        if (row < n) csum += __expf(acc[r] * scale - mArr[row]) * zArr[row];
      }
    }
    csum += __shfl_xor(csum, 16);          // fold rows 0-7 half with 8-15 half
    const int col = tj * 16 + l15;
    if (hi == 0 && col < n) cArr[col] = csum;
  }
  __syncthreads();

  // ---- out[g][f] = sum_j c_j * V[s+j][f]  (fp32, fully coalesced)
  float accf = 0.f;
  for (int j = 0; j < n; ++j)
    accf = fmaf(cArr[j], Vf[(size_t)(s + j) * N_FEAT + tid], accf);
  out[(size_t)g * N_FEAT + tid] = accf;
}

// ---------------------------------------------------------------------------
// Host launcher
// ---------------------------------------------------------------------------
extern "C" void kernel_launch(void* const* d_in, const int* in_sizes, int n_in,
                              void* d_out, int out_size, void* d_ws, size_t ws_size,
                              hipStream_t stream) {
  (void)in_sizes; (void)n_in; (void)out_size; (void)ws_size;
  const float* X     = (const float*)d_in[0];
  const int*   batch = (const int*)d_in[1];
  const float* Wq    = (const float*)d_in[2];
  const float* Wk    = (const float*)d_in[3];
  const float* Wv    = (const float*)d_in[4];
  float*       out   = (float*)d_out;

  char* ws = (char*)d_ws;
  _Float16* WhT = (_Float16*)(ws + 0);                 //  512*256*2 =   262,144
  _Float16* Qh  = (_Float16*)(ws + 262144);            // 12288*128*2 = 3,145,728
  _Float16* Kh  = (_Float16*)(ws + 3407872);           // 12288*128*2 = 3,145,728
  float*    Vf  = (float*)   (ws + 6553600);           // 12288*256*4 = 12,582,912

  k_build_wt<<<dim3(NCOLS), dim3(256), 0, stream>>>(Wq, Wk, Wv, WhT);
  k_proj_gemm<<<dim3(N_NODES / 128, NCOLS / 128), dim3(256), 0, stream>>>(X, WhT, Qh, Kh, Vf);
  k_attn_pool<<<dim3(N_GRAPHS), dim3(256), 0, stream>>>(Qh, Kh, Vf, batch, out);
}
